// HebbianAttention_51573967291051
// MI455X (gfx1250) — matrix operand order repaired and checked by reading, not compiled, not run on hardware
//
#include <hip/hip_runtime.h>

// MI455X / gfx1250, wave32. All matrix work through v_wmma_f32_16x16x32_bf16.
typedef __attribute__((ext_vector_type(16))) __bf16 v16bf;
typedef __attribute__((ext_vector_type(8)))  __bf16 v8bf;
typedef __attribute__((ext_vector_type(8)))  float  v8f;

#define WMMA_BF16(a, b, c) \
  __builtin_amdgcn_wmma_f32_16x16x32_bf16(false, (a), false, (b), (short)0, (c), false, false)

constexpr int Bc = 2, Sc = 2048, Dc = 1024, Hc = 16, DKc = 64, EDc = 256, TOPKc = 32;

// Build a 16-elem bf16 fragment from 16 contiguous elements (two aligned v8bf).
__device__ inline v16bf frag16(const __bf16* p) {
  v8bf lo = *(const v8bf*)p;
  v8bf hi = *(const v8bf*)(p + 8);
  return __builtin_shufflevector(lo, hi, 0,1,2,3,4,5,6,7,8,9,10,11,12,13,14,15);
}
// A-fragment: elems [0..7] at p, [8..15] at p+16 (ISA 16-bit A layout).
__device__ inline v16bf fragA(const __bf16* p) {
  v8bf lo = *(const v8bf*)p;
  v8bf hi = *(const v8bf*)(p + 16);
  return __builtin_shufflevector(lo, hi, 0,1,2,3,4,5,6,7,8,9,10,11,12,13,14,15);
}
__device__ inline unsigned int pack2f(float a, float b) {
  union { __bf16 h[2]; unsigned int u; } p;
  p.h[0] = (__bf16)a; p.h[1] = (__bf16)b; return p.u;
}
__device__ inline unsigned int pack2h(__bf16 a, __bf16 b) {
  union { __bf16 h[2]; unsigned int u; } p;
  p.h[0] = a; p.h[1] = b; return p.u;
}

// ---------------------------------------------------------------------------
// Kernel 1: per-head effective alphas:  a = 0.05 / (1 + ||T_h||_F / 1.0)
// ---------------------------------------------------------------------------
__global__ __launch_bounds__(256) void k_alpha(const float* __restrict__ tr,
                                               const float* __restrict__ vtr,
                                               float* __restrict__ aT,
                                               float* __restrict__ aV) {
  __shared__ float red[256];
  int h = blockIdx.x, t = threadIdx.x;
  float s1 = 0.f;
  for (int i = t; i < DKc * DKc; i += 256) { float v = tr[h * DKc * DKc + i]; s1 += v * v; }
  red[t] = s1; __syncthreads();
  for (int o = 128; o > 0; o >>= 1) { if (t < o) red[t] += red[t + o]; __syncthreads(); }
  if (t == 0) aT[h] = 0.05f / (1.f + sqrtf(red[0]));
  __syncthreads();
  float s2 = 0.f;
  for (int i = t; i < EDc * DKc; i += 256) { float v = vtr[h * EDc * DKc + i]; s2 += v * v; }
  red[t] = s2; __syncthreads();
  for (int o = 128; o > 0; o >>= 1) { if (t < o) red[t] += red[t + o]; __syncthreads(); }
  if (t == 0) aV[h] = 0.05f / (1.f + sqrtf(red[0]));
}

// ---------------------------------------------------------------------------
// Kernel 2: fused QKV projection.  P = X(4096x1024) @ W(1024x1024), W picked
// by blockIdx.z.  Output bf16 in (B,H,S,DK).  Block = 128 thr (4 waves),
// 64x64 output tile, K staged 64 at a time; W staged TRANSPOSED so B-frags
// are contiguous v8bf loads.  All frag loads hoisted ahead of the WMMA block.
// ---------------------------------------------------------------------------
__global__ __launch_bounds__(128) void k_proj(const float* __restrict__ x,
                                              const float* __restrict__ Wq,
                                              const float* __restrict__ Wk,
                                              const float* __restrict__ Wv,
                                              __bf16* __restrict__ Qb,
                                              __bf16* __restrict__ Kb,
                                              __bf16* __restrict__ Vb) {
  const float* W = (blockIdx.z == 0) ? Wq : (blockIdx.z == 1) ? Wk : Wv;
  __bf16* dst    = (blockIdx.z == 0) ? Qb : (blockIdx.z == 1) ? Kb : Vb;
  __shared__ __bf16 As[64][72];     // [row][k]   stride 144B (16B aligned)
  __shared__ __bf16 Bst[64][72];    // [col][k]   transposed W tile
  int tid = threadIdx.x, lane = tid & 31, w = tid >> 5;
  int rowT = blockIdx.x * 64, colT = blockIdx.y * 64;
  int m = lane & 15, hh = lane >> 4, n = lane & 15;
  v8f acc[4];
  for (int c = 0; c < 4; c++) for (int r = 0; r < 8; r++) acc[c][r] = 0.f;
  for (int k0 = 0; k0 < Dc; k0 += 64) {
    { int r = tid >> 1, c0 = (tid & 1) * 32;          // A: 64 rows x 64 k
      const float* src = &x[(rowT + r) * Dc + k0 + c0];
      for (int i = 0; i < 32; i++) As[r][c0 + i] = (__bf16)src[i]; }
    { int rp = tid >> 2, cg = tid & 3;                // B: transpose 2k x 16col
      int r = rp * 2, c0 = cg * 16;
      const float* s0 = &W[(k0 + r) * Dc + colT + c0];
      const float* s1 = s0 + Dc;
      for (int i = 0; i < 16; i++)
        *(unsigned int*)&Bst[c0 + i][r] = pack2f(s0[i], s1[i]); }
    if (k0 + 64 < Dc) __builtin_prefetch(&W[(k0 + 64 + (tid >> 2) * 2) * Dc + colT], 0, 1);
    __syncthreads();
    // hoist all fragment loads, then issue the 8 WMMAs densely
    v16bf a0 = fragA(&As[w * 16 + m][hh * 8]);
    v16bf a1 = fragA(&As[w * 16 + m][32 + hh * 8]);
    v16bf b0[4], b1[4];
    for (int c = 0; c < 4; c++) {
      b0[c] = fragA(&Bst[c * 16 + n][hh * 8]);
      b1[c] = fragA(&Bst[c * 16 + n][32 + hh * 8]);
    }
    for (int c = 0; c < 4; c++) acc[c] = WMMA_BF16(a0, b0[c], acc[c]);
    for (int c = 0; c < 4; c++) acc[c] = WMMA_BF16(a1, b1[c], acc[c]);
    __syncthreads();
  }
  for (int c = 0; c < 4; c++)
    for (int r = 0; r < 8; r++) {
      int row = rowT + w * 16 + r + 8 * hh;       // row = b*S + s
      int col = colT + c * 16 + n;                // col = h*64 + dk
      int b = row >> 11, s = row & 2047, hd = col >> 6, dk = col & 63;
      dst[(((size_t)b * Hc + hd) * Sc + s) * DKc + dk] = (__bf16)acc[c][r];
    }
}

// ---------------------------------------------------------------------------
// Kernel 3: Qeff = Q @ (I + alphaT[h] * T_h).  Trace-score path folded into
// Q so attention needs a single S x S GEMM.  M staged transposed in LDS.
// ---------------------------------------------------------------------------
__global__ __launch_bounds__(128) void k_qeff(const __bf16* __restrict__ Qb,
                                              const float* __restrict__ tr,
                                              const float* __restrict__ aT,
                                              __bf16* __restrict__ Qe) {
  __shared__ __bf16 Mst[64][72];    // [e][d] = (I + a*T)^T, so B-frags contiguous
  int tid = threadIdx.x, lane = tid & 31, w = tid >> 5;
  int h = blockIdx.y, b = blockIdx.z;
  int m = lane & 15, hh = lane >> 4, n = m;
  float a_h = aT[h];
  { int rp = tid >> 2, cg = tid & 3;
    int r = rp * 2, c0 = cg * 16;                 // rows d=r,r+1; cols e=c0..c0+15
    const float* s0 = &tr[(h * DKc + r) * DKc + c0];
    const float* s1 = s0 + DKc;
    for (int i = 0; i < 16; i++) {
      float v0 = s0[i] * a_h + ((r     == c0 + i) ? 1.f : 0.f);
      float v1 = s1[i] * a_h + ((r + 1 == c0 + i) ? 1.f : 0.f);
      *(unsigned int*)&Mst[c0 + i][r] = pack2f(v0, v1);
    } }
  __syncthreads();
  int sb = blockIdx.x * 64 + w * 16;
  const __bf16* qr = &Qb[((size_t)(b * Hc + h) * Sc + sb + m) * DKc];
  v8f acc[4];
  for (int c = 0; c < 4; c++) for (int r = 0; r < 8; r++) acc[c][r] = 0.f;
  v16bf a0 = fragA(&qr[hh * 8]);
  v16bf a1 = fragA(&qr[32 + hh * 8]);
  v16bf b0[4], b1[4];
  for (int c = 0; c < 4; c++) {
    b0[c] = fragA(&Mst[c * 16 + n][hh * 8]);
    b1[c] = fragA(&Mst[c * 16 + n][32 + hh * 8]);
  }
  for (int c = 0; c < 4; c++) acc[c] = WMMA_BF16(a0, b0[c], acc[c]);
  for (int c = 0; c < 4; c++) acc[c] = WMMA_BF16(a1, b1[c], acc[c]);
  for (int c = 0; c < 4; c++)
    for (int r = 0; r < 8; r++) {
      int s = sb + r + 8 * hh, col = c * 16 + n;
      Qe[((size_t)(b * Hc + h) * Sc + s) * DKc + col] = (__bf16)acc[c][r];
    }
}

// ---------------------------------------------------------------------------
// Kernel 4: pattern-separation value-trace read.  One wave per (b,h,s) row:
//   e = relu(Q[s-1] @ W_expand);  kth = 32nd largest (wave argmax x32);
//   O = alphaV[h] * (e>=kth ? e : 0) @ value_traces[h]
// O is fully written here; attention adds on top afterwards.
// ---------------------------------------------------------------------------
__global__ __launch_bounds__(256) void k_expand(const __bf16* __restrict__ Qb,
                                                const float* __restrict__ Wex,
                                                const float* __restrict__ vtr,
                                                const float* __restrict__ aV,
                                                float* __restrict__ O) {
  __shared__ float qsh[8][64];
  __shared__ float esh[8][256];
  int tid = threadIdx.x, lane = tid & 31, w = tid >> 5;
  int gid = blockIdx.x * 8 + w;          // gid = (b*H + h)*S + s
  int s = gid & (Sc - 1);
  int h = (gid >> 11) & (Hc - 1);
  int b = gid >> 15;
  const int bh = b * Hc + h;
  if (s == 0) { qsh[w][lane] = 0.f; qsh[w][lane + 32] = 0.f; }
  else {
    const __bf16* q = &Qb[((size_t)bh * Sc + (s - 1)) * DKc];
    qsh[w][lane] = (float)q[lane];
    qsh[w][lane + 32] = (float)q[lane + 32];
  }
  __syncthreads();
  float e[8];
  int c0 = lane * 8;
  for (int j = 0; j < 8; j++) e[j] = 0.f;
  for (int k = 0; k < DKc; k++) {
    float qk = qsh[w][k];
    const float* wr = &Wex[k * EDc + c0];
    for (int j = 0; j < 8; j++) e[j] += qk * wr[j];
  }
  for (int j = 0; j < 8; j++) e[j] = e[j] > 0.f ? e[j] : 0.f;
  float wv[8];
  for (int j = 0; j < 8; j++) wv[j] = e[j];
  float kth = 0.f;
  for (int t = 0; t < TOPKc; t++) {
    float lm = -1.f;
    for (int j = 0; j < 8; j++) lm = wv[j] > lm ? wv[j] : lm;
    int owner = lane;
    for (int off = 16; off >= 1; off >>= 1) {
      float ov = __shfl_xor(lm, off);
      int   oo = __shfl_xor(owner, off);
      if (ov > lm || (ov == lm && oo < owner)) { lm = ov; owner = oo; }
    }
    kth = lm;
    if (lane == owner) {
      for (int j = 0; j < 8; j++) if (wv[j] == lm) { wv[j] = -2.f; break; }
    }
  }
  for (int j = 0; j < 8; j++) esh[w][c0 + j] = (e[j] >= kth) ? e[j] : 0.f;
  __syncthreads();
  float s0 = 0.f, s1 = 0.f;
  int d0 = lane, d1 = lane + 32;
  for (int c = 0; c < EDc; c++) {
    float ev = esh[w][c];
    if (ev != 0.f) {
      const float* vt = &vtr[((size_t)h * EDc + c) * DKc];
      s0 += ev * vt[d0];
      s1 += ev * vt[d1];
    }
  }
  float av = aV[h];
  size_t oi = ((size_t)bh * Sc + s) * DKc;
  O[oi + d0] = av * s0;
  O[oi + d1] = av * s1;
}

// ---------------------------------------------------------------------------
// Kernel 5: flash attention per (b,h).  Block = 4 waves x 16 query rows,
// 64-key tiles.  K staged row-major, V staged TRANSPOSED [feat][key] so all
// WMMA B-frags are contiguous.  16 WMMA per barrier pair (8 score + 8 A@V),
// with all fragment loads hoisted ahead of each WMMA block.
// ---------------------------------------------------------------------------
__global__ __launch_bounds__(128) void k_attn(const __bf16* __restrict__ Qe,
                                              const __bf16* __restrict__ Kb,
                                              const __bf16* __restrict__ Vb,
                                              float* __restrict__ O) {
  __shared__ __bf16 Ks[64][72];     // [key][feat]
  __shared__ __bf16 Vts[64][72];    // [feat][key]  (transposed V tile)
  __shared__ __bf16 Ps[4][16][72];  // per-wave P 16x64
  int tid = threadIdx.x, lane = tid & 31, w = tid >> 5;
  int h = blockIdx.y, b = blockIdx.z;
  const size_t bh = (size_t)(b * Hc + h);
  int m = lane & 15, hh = lane >> 4, n = m;
  int qb0 = blockIdx.x * 64 + w * 16;
  const __bf16* qr = &Qe[(bh * Sc + qb0 + m) * DKc];
  v16bf a_lo = fragA(&qr[hh * 8]);
  v16bf a_hi = fragA(&qr[32 + hh * 8]);
  float mrow[8], lrow[8];
  for (int r = 0; r < 8; r++) { mrow[r] = -1e30f; lrow[r] = 0.f; }
  v8f acc[4];
  for (int c = 0; c < 4; c++) for (int r = 0; r < 8; r++) acc[c][r] = 0.f;
  const float scale = 0.125f;  // 1/sqrt(64)
  const __bf16* Kg = &Kb[bh * Sc * DKc];
  const __bf16* Vg = &Vb[bh * Sc * DKc];
  for (int kb = 0; kb < Sc; kb += 64) {
    { int r = tid >> 1, c0 = (tid & 1) * 32;          // K copy: row r, 32 feats
      const v8bf* src = (const v8bf*)&Kg[(size_t)(kb + r) * DKc + c0];
      v8bf* dstp = (v8bf*)&Ks[r][c0];
      dstp[0] = src[0]; dstp[1] = src[1]; dstp[2] = src[2]; dstp[3] = src[3]; }
    { int kp = tid >> 2, fg = tid & 3;                // V transpose: 2 keys x 16 feats
      int k = kp * 2, f0 = fg * 16;
      const __bf16* v0 = &Vg[(size_t)(kb + k) * DKc + f0];
      const __bf16* v1 = v0 + DKc;
      for (int i = 0; i < 16; i++)
        *(unsigned int*)&Vts[f0 + i][k] = pack2h(v0[i], v1[i]); }
    if (kb + 64 < Sc) {
      __builtin_prefetch(&Kg[(size_t)(kb + 64 + (tid >> 1)) * DKc], 0, 1);
      __builtin_prefetch(&Vg[(size_t)(kb + 64 + (tid >> 1)) * DKc], 0, 1);
    }
    __syncthreads();
    // ---- scores: load all 8 K-frags, then 8 WMMAs ----
    v16bf bk0[4], bk1[4];
    for (int g = 0; g < 4; g++) {
      int key = g * 16 + n;
      bk0[g] = fragA(&Ks[key][hh * 8]);
      bk1[g] = fragA(&Ks[key][32 + hh * 8]);
    }
    v8f sc[4];
    for (int g = 0; g < 4; g++) for (int r = 0; r < 8; r++) sc[g][r] = 0.f;
    for (int g = 0; g < 4; g++) sc[g] = WMMA_BF16(a_lo, bk0[g], sc[g]);
    for (int g = 0; g < 4; g++) sc[g] = WMMA_BF16(a_hi, bk1[g], sc[g]);
    // ---- online softmax (rows M=r / r+8 per lane half) ----
    float corr[8];
    for (int r = 0; r < 8; r++) {
      float v = sc[0][r];
      v = sc[1][r] > v ? sc[1][r] : v;
      v = sc[2][r] > v ? sc[2][r] : v;
      v = sc[3][r] > v ? sc[3][r] : v;
      for (int off = 1; off < 16; off <<= 1) { float o = __shfl_xor(v, off); v = o > v ? o : v; }
      float mn = v * scale;
      mn = mrow[r] > mn ? mrow[r] : mn;
      corr[r] = __expf(mrow[r] - mn);
      mrow[r] = mn;
      float ps = 0.f;
      for (int g = 0; g < 4; g++) {
        float p = __expf(sc[g][r] * scale - mn);
        ps += p;
        Ps[w][r + 8 * hh][g * 16 + n] = (__bf16)p;
      }
      for (int off = 1; off < 16; off <<= 1) ps += __shfl_xor(ps, off);
      lrow[r] = lrow[r] * corr[r] + ps;
    }
    for (int c = 0; c < 4; c++) for (int r = 0; r < 8; r++) acc[c][r] *= corr[r];
    __syncthreads();
    // ---- A @ V: load P-frags + all 8 V-frags, then 8 WMMAs ----
    v16bf pa0 = fragA(&Ps[w][m][hh * 8]);        // keys 0..31
    v16bf pa1 = fragA(&Ps[w][m][32 + hh * 8]);   // keys 32..63
    v16bf bv0[4], bv1[4];
    for (int c = 0; c < 4; c++) {
      bv0[c] = fragA(&Vts[c * 16 + n][hh * 8]);
      bv1[c] = fragA(&Vts[c * 16 + n][32 + hh * 8]);
    }
    for (int c = 0; c < 4; c++) acc[c] = WMMA_BF16(pa0, bv0[c], acc[c]);
    for (int c = 0; c < 4; c++) acc[c] = WMMA_BF16(pa1, bv1[c], acc[c]);
    __syncthreads();
  }
  for (int c = 0; c < 4; c++)
    for (int r = 0; r < 8; r++) {
      int s = qb0 + r + 8 * hh, col = c * 16 + n;
      size_t oi = (bh * Sc + s) * DKc + col;
      O[oi] += acc[c][r] / lrow[r];
    }
}

// ---------------------------------------------------------------------------
// Kernel 6: output projection Y = O_(B,S,D-view) @ W_o, with the
// (B,H,S,DK) -> (B,S,D) remap done while staging A into LDS.
// ---------------------------------------------------------------------------
__global__ __launch_bounds__(128) void k_oproj(const float* __restrict__ O,
                                               const float* __restrict__ Wo,
                                               float* __restrict__ Y) {
  __shared__ __bf16 As[64][72];
  __shared__ __bf16 Bst[64][72];    // [col][k] transposed W_o tile
  int tid = threadIdx.x, lane = tid & 31, w = tid >> 5;
  int rowT = blockIdx.x * 64, colT = blockIdx.y * 64;
  int m = lane & 15, hh = lane >> 4, n = m;
  v8f acc[4];
  for (int c = 0; c < 4; c++) for (int r = 0; r < 8; r++) acc[c][r] = 0.f;
  for (int k0 = 0; k0 < Dc; k0 += 64) {
    { int r = tid >> 1, cc = (tid & 1) * 32;
      int grow = rowT + r;
      int bb = grow >> 11, ss = grow & 2047;
      for (int i = 0; i < 32; i++) {
        int gcol = k0 + cc + i;
        int hd = gcol >> 6, dk = gcol & 63;
        As[r][cc + i] = (__bf16)O[(((size_t)bb * Hc + hd) * Sc + ss) * DKc + dk];
      } }
    { int rp = tid >> 2, cg = tid & 3;
      int r = rp * 2, cc = cg * 16;
      const float* s0 = &Wo[(k0 + r) * Dc + colT + cc];
      const float* s1 = s0 + Dc;
      for (int i = 0; i < 16; i++)
        *(unsigned int*)&Bst[cc + i][r] = pack2f(s0[i], s1[i]); }
    __syncthreads();
    v16bf a0 = fragA(&As[w * 16 + m][hh * 8]);
    v16bf a1 = fragA(&As[w * 16 + m][32 + hh * 8]);
    v16bf b0[4], b1[4];
    for (int c = 0; c < 4; c++) {
      b0[c] = fragA(&Bst[c * 16 + n][hh * 8]);
      b1[c] = fragA(&Bst[c * 16 + n][32 + hh * 8]);
    }
    for (int c = 0; c < 4; c++) acc[c] = WMMA_BF16(a0, b0[c], acc[c]);
    for (int c = 0; c < 4; c++) acc[c] = WMMA_BF16(a1, b1[c], acc[c]);
    __syncthreads();
  }
  for (int c = 0; c < 4; c++)
    for (int r = 0; r < 8; r++) {
      int row = rowT + w * 16 + r + 8 * hh, col = colT + c * 16 + n;
      Y[(size_t)row * Dc + col] = acc[c][r];
    }
}

// ---------------------------------------------------------------------------
extern "C" void kernel_launch(void* const* d_in, const int* in_sizes, int n_in,
                              void* d_out, int out_size, void* d_ws, size_t ws_size,
                              hipStream_t stream) {
  const float* x   = (const float*)d_in[0];
  const float* Wq  = (const float*)d_in[1];
  const float* Wk  = (const float*)d_in[2];
  const float* Wv  = (const float*)d_in[3];
  const float* Wo  = (const float*)d_in[4];
  const float* tr  = (const float*)d_in[5];
  const float* vtr = (const float*)d_in[6];
  const float* Wex = (const float*)d_in[7];

  char* ws = (char*)d_ws;
  const size_t NE = (size_t)Bc * Hc * Sc * DKc;      // 4 Mi elements
  __bf16* Qb = (__bf16*)(ws);                         // 8 MiB
  __bf16* Kb = (__bf16*)(ws + NE * 2);                // 8 MiB
  __bf16* Vb = (__bf16*)(ws + NE * 4);                // 8 MiB
  __bf16* Qe = (__bf16*)(ws + NE * 6);                // 8 MiB
  float*  O  = (float*)(ws + NE * 8);                 // 16 MiB
  float*  aT = (float*)(ws + NE * 8 + NE * 4);        // 16 floats
  float*  aV = aT + Hc;                               // 16 floats

  k_alpha <<<Hc, 256, 0, stream>>>(tr, vtr, aT, aV);
  k_proj  <<<dim3(64, 16, 3), 128, 0, stream>>>(x, Wq, Wk, Wv, Qb, Kb, Vb);
  k_qeff  <<<dim3(32, 16, 2), 128, 0, stream>>>(Qb, tr, aT, Qe);
  k_expand<<<(Bc * Hc * Sc) / 8, 256, 0, stream>>>(Qb, Wex, vtr, aV, O);
  k_attn  <<<dim3(32, 16, 2), 128, 0, stream>>>(Qe, Kb, Vb, O);
  k_oproj <<<dim3(64, 16), 128, 0, stream>>>(O, Wo, (float*)d_out);
}
